// DecoderLayer_33157147525831
// MI455X (gfx1250) — matrix operand order repaired and checked by reading, not compiled, not run on hardware
//
#include <hip/hip_runtime.h>
#include <hip/hip_bf16.h>

// ---------------- problem constants ----------------
#define B_     4
#define SQ_    1024
#define SK_    1024
#define D_     1024
#define H_     16
#define DEPTH_ 64
#define DFF_   4096

typedef __attribute__((ext_vector_type(16))) __bf16 v16bf;
typedef __attribute__((ext_vector_type(8)))  __bf16 v8bf;
typedef __attribute__((ext_vector_type(4)))  __bf16 v4bf;
typedef __attribute__((ext_vector_type(8)))  float  v8f;
typedef __attribute__((ext_vector_type(4)))  int    v4i;

__device__ __forceinline__ __bf16 f2bf(float f) { return (__bf16)f; }

// LDS row stride: 40 bf16 = 80 B = 20 banks -> conflict-free fragment reads,
// all v8bf/v4bf LDS accesses stay 8/16-byte aligned.
#define LDSW 40
#define BUFE (128 * LDSW)   // elements per LDS buffer

// ---- CDNA5 async global->LDS copy (no VGPR data path, ASYNCcnt) ----
__device__ __forceinline__ void async_cp16(const __bf16* g, __bf16* l) {
#if __has_builtin(__builtin_amdgcn_global_load_async_to_lds_b128)
  __bf16* gnc = const_cast<__bf16*>(g);
  __builtin_amdgcn_global_load_async_to_lds_b128(
      (__attribute__((address_space(1))) v4i*)gnc,
      (__attribute__((address_space(3))) v4i*)l, 0, 0);
#else
  unsigned lofs = (unsigned)(size_t)(__attribute__((address_space(3))) void*)l;
  asm volatile("global_load_async_to_lds_b128 %0, %1, off"
               :: "v"(lofs), "v"((unsigned long long)(size_t)g) : "memory");
#endif
}
__device__ __forceinline__ void wait_async0() {
#if __has_builtin(__builtin_amdgcn_s_wait_asynccnt)
  __builtin_amdgcn_s_wait_asynccnt(0);
#else
  asm volatile("s_wait_asynccnt 0x0" ::: "memory");
#endif
}

// =====================================================================
// Tiled WMMA GEMM:  C = act(alpha * A@B + bias) [+ Res]
//   A: M x K fp32 row-major.
//   B: BMODE 0: K x N fp32 | 1: N x K fp32 | 2: N x K bf16 (packed)
//   BMODE 2 stages B via async global->LDS copies (zero VGPR traffic).
//   NG: apply N bounds guard (only attn@V, N=64)
// Block 128x128, K-step 32, 8 waves (4x2), wave tile 32x64.
// LDS double buffered; 1 barrier per K-step; global latency hides
// under the 8 WMMAs.
// =====================================================================
template<int ACT, int BMODE, bool RES, bool NG, bool BIAS>
__global__ __launch_bounds__(256)
void gemm_wmma_kernel(const float* __restrict__ A, const void* __restrict__ Bvp,
                      const float* __restrict__ bias, const float* __restrict__ Res,
                      float* __restrict__ C,
                      int M, int N, int K,
                      int lda, int ldb, int ldc, int ldres,
                      int nheads,
                      long long sAb, long long sAh,
                      long long sBb, long long sBh,
                      long long sCb, long long sCh,
                      float alpha)
{
  __shared__ __bf16 As[2 * BUFE];
  __shared__ __bf16 Bs[2 * BUFE];

  const int tid = threadIdx.x;
  const int z   = blockIdx.z;
  const int bb  = z / nheads;
  const int hh  = z - bb * nheads;
  const float* Ap = A + (long long)bb * sAb + (long long)hh * sAh;
  float*       Cp = C + (long long)bb * sCb + (long long)hh * sCh;
  const long long offB = (long long)bb * sBb + (long long)hh * sBh;
  const float*  Bpf = (BMODE != 2) ? ((const float*)Bvp + offB) : nullptr;
  const __bf16* Bp8 = (BMODE == 2) ? ((const __bf16*)Bvp + offB) : nullptr;

  const int blockM = blockIdx.y * 128;
  const int blockN = blockIdx.x * 128;

  const int wave  = tid >> 5;
  const int lane  = tid & 31;
  const int wr    = wave & 3;
  const int wc    = wave >> 2;
  const int lrow  = lane & 15;
  const int lhalf = lane >> 4;

  // ---- precomputed 32-bit per-thread offsets ----
  int agoff[4], asoff[4];
#pragma unroll
  for (int it = 0; it < 4; ++it) {
    int idx = tid + (it << 8);
    int r   = idx >> 3;
    int c4  = (idx & 7) << 2;
    agoff[it] = (blockM + r) * lda + c4;
    asoff[it] = r * LDSW + c4;
  }
  int bgoff[4], bsoff[4];
  bool bok[4] = {true, true, true, true};
  if (BMODE == 2) {
#pragma unroll
    for (int it = 0; it < 2; ++it) {
      int idx = tid + (it << 8);           // 512 x (8 bf16)
      int nr  = idx >> 2;
      int c8  = (idx & 3) << 3;
      bgoff[it] = (blockN + nr) * ldb + c8;
      bsoff[it] = nr * LDSW + c8;
    }
  } else if (BMODE == 1) {
#pragma unroll
    for (int it = 0; it < 4; ++it) {
      int idx = tid + (it << 8);
      int nr  = idx >> 3;
      int c4  = (idx & 7) << 2;
      bgoff[it] = (blockN + nr) * ldb + c4;
      bsoff[it] = nr * LDSW + c4;
    }
  } else {
#pragma unroll
    for (int it = 0; it < 4; ++it) {
      int idx = tid + (it << 8);
      int kr  = idx >> 5;
      int nc4 = (idx & 31) << 2;
      bgoff[it] = kr * ldb + blockN + nc4;
      bsoff[it] = nc4 * LDSW + kr;
      if (NG) bok[it] = (blockN + nc4) < N;
    }
  }
  int afoff[2];
#pragma unroll
  for (int mi = 0; mi < 2; ++mi)
    afoff[mi] = (wr * 32 + mi * 16 + lrow) * LDSW + lhalf * 8;
  int bfoff[4];
#pragma unroll
  for (int ni = 0; ni < 4; ++ni)
    bfoff[ni] = (wc * 64 + ni * 16 + lrow) * LDSW + lhalf * 8;

  v8f acc[2][4];
#pragma unroll
  for (int mi = 0; mi < 2; ++mi)
#pragma unroll
    for (int ni = 0; ni < 4; ++ni)
      acc[mi][ni] = (v8f){0.f,0.f,0.f,0.f,0.f,0.f,0.f,0.f};

  float4 ra[4];
  float4 rbf[4];

  const int ksteps = K >> 5;

  // ---- prologue: stage tile 0 into buffer 0 ----
  {
    if (BMODE == 2) {
#pragma unroll
      for (int it = 0; it < 2; ++it)
        async_cp16(Bp8 + bgoff[it], &Bs[bsoff[it]]);
    } else if (BMODE == 1) {
#pragma unroll
      for (int it = 0; it < 4; ++it) rbf[it] = *(const float4*)(Bpf + bgoff[it]);
    } else {
#pragma unroll
      for (int it = 0; it < 4; ++it) {
        float4 v = make_float4(0.f, 0.f, 0.f, 0.f);
        if (bok[it]) v = *(const float4*)(Bpf + bgoff[it]);
        rbf[it] = v;
      }
    }
#pragma unroll
    for (int it = 0; it < 4; ++it) ra[it] = *(const float4*)(Ap + agoff[it]);
#pragma unroll
    for (int it = 0; it < 4; ++it) {
      v4bf pk = { f2bf(ra[it].x), f2bf(ra[it].y), f2bf(ra[it].z), f2bf(ra[it].w) };
      *(v4bf*)&As[asoff[it]] = pk;
    }
    if (BMODE == 1) {
#pragma unroll
      for (int it = 0; it < 4; ++it) {
        v4bf pk = { f2bf(rbf[it].x), f2bf(rbf[it].y), f2bf(rbf[it].z), f2bf(rbf[it].w) };
        *(v4bf*)&Bs[bsoff[it]] = pk;
      }
    } else if (BMODE == 0) {
#pragma unroll
      for (int it = 0; it < 4; ++it) {
        Bs[bsoff[it] + 0 * LDSW] = f2bf(rbf[it].x);
        Bs[bsoff[it] + 1 * LDSW] = f2bf(rbf[it].y);
        Bs[bsoff[it] + 2 * LDSW] = f2bf(rbf[it].z);
        Bs[bsoff[it] + 3 * LDSW] = f2bf(rbf[it].w);
      }
    }
    if (BMODE == 2) wait_async0();
  }
  __syncthreads();

  int bufo = 0;
  for (int s = 0; s < ksteps; ++s) {
    const bool hasNext = (s + 1 < ksteps);
    const int  nb = bufo ^ BUFE;

    // B (packed bf16): kick off async copies for next tile as early as
    // possible — they bypass VGPRs entirely and complete under the WMMAs.
    if (hasNext && BMODE == 2) {
      const int k1 = (s + 1) << 5;
#pragma unroll
      for (int it = 0; it < 2; ++it)
        async_cp16(Bp8 + k1 + bgoff[it], &Bs[nb + bsoff[it]]);
    }

    // ---- fragments from LDS[cur] (ISA VGPR layouts) ----
    v16bf afrag[2];
#pragma unroll
    for (int mi = 0; mi < 2; ++mi) {
      v8bf a0 = *(const v8bf*)&As[bufo + afoff[mi]];
      v8bf a1 = *(const v8bf*)&As[bufo + afoff[mi] + 16];
      v16bf a;
#pragma unroll
      for (int j = 0; j < 8; ++j) { a[j] = a0[j]; a[j + 8] = a1[j]; }
      afrag[mi] = a;
    }
    v16bf bfrag[4];
#pragma unroll
    for (int ni = 0; ni < 4; ++ni) {
      v8bf b0 = *(const v8bf*)&Bs[bufo + bfoff[ni]];
      v8bf b1 = *(const v8bf*)&Bs[bufo + bfoff[ni] + 16];
      v16bf b;
#pragma unroll
      for (int j = 0; j < 8; ++j) { b[j] = b0[j]; b[j + 8] = b1[j]; }
      bfrag[ni] = b;
    }

    if (hasNext) {
      // issue next A (and fp32-B) global loads; latency hides under WMMAs
      const int k1 = (s + 1) << 5;
      const float* ApK = Ap + k1;
#pragma unroll
      for (int it = 0; it < 4; ++it) ra[it] = *(const float4*)(ApK + agoff[it]);
      if (BMODE == 1) {
        const float* BfK = Bpf + k1;
#pragma unroll
        for (int it = 0; it < 4; ++it) rbf[it] = *(const float4*)(BfK + bgoff[it]);
      } else if (BMODE == 0) {
        const float* BfK = Bpf + (size_t)k1 * ldb;
#pragma unroll
        for (int it = 0; it < 4; ++it) {
          float4 v = make_float4(0.f, 0.f, 0.f, 0.f);
          if (bok[it]) v = *(const float4*)(BfK + bgoff[it]);
          rbf[it] = v;
        }
      }
    }

#pragma unroll
    for (int mi = 0; mi < 2; ++mi)
#pragma unroll
      for (int ni = 0; ni < 4; ++ni)
        acc[mi][ni] = __builtin_amdgcn_wmma_f32_16x16x32_bf16(
            false, afrag[mi], false, bfrag[ni],
            (short)0, acc[mi][ni], false, false);

    if (hasNext) {
#pragma unroll
      for (int it = 0; it < 4; ++it) {
        v4bf pk = { f2bf(ra[it].x), f2bf(ra[it].y), f2bf(ra[it].z), f2bf(ra[it].w) };
        *(v4bf*)&As[nb + asoff[it]] = pk;
      }
      if (BMODE == 1) {
#pragma unroll
        for (int it = 0; it < 4; ++it) {
          v4bf pk = { f2bf(rbf[it].x), f2bf(rbf[it].y), f2bf(rbf[it].z), f2bf(rbf[it].w) };
          *(v4bf*)&Bs[nb + bsoff[it]] = pk;
        }
      } else if (BMODE == 0) {
#pragma unroll
        for (int it = 0; it < 4; ++it) {
          Bs[nb + bsoff[it] + 0 * LDSW] = f2bf(rbf[it].x);
          Bs[nb + bsoff[it] + 1 * LDSW] = f2bf(rbf[it].y);
          Bs[nb + bsoff[it] + 2 * LDSW] = f2bf(rbf[it].z);
          Bs[nb + bsoff[it] + 3 * LDSW] = f2bf(rbf[it].w);
        }
      } else {
        wait_async0();                 // own async B copies landed in LDS
      }
      __syncthreads();
    }
    bufo ^= BUFE;
  }

  // ---- epilogue ----
  const int crow0 = blockM + wr * 32 + lhalf * 8;
  const int ccol0 = blockN + wc * 64 + lrow;
#pragma unroll
  for (int mi = 0; mi < 2; ++mi)
#pragma unroll
    for (int ni = 0; ni < 4; ++ni)
#pragma unroll
      for (int j = 0; j < 8; ++j) {
        int row = crow0 + mi * 16 + j;
        int col = ccol0 + ni * 16;
        if (!NG || col < N) {
          float v = acc[mi][ni][j] * alpha;
          if (BIAS) v += bias[col];
          if (ACT == 1) v = v > 0.f ? v : 0.f;
          if (RES) v += Res[row * ldres + col];
          Cp[row * ldc + col] = v;
        }
      }
}

// =====================================================================
// Weight pack: fp32 [K][N] row-major -> bf16 [N][K] row-major
// =====================================================================
__global__ __launch_bounds__(256)
void transpose_pack_kernel(const float* __restrict__ in, __bf16* __restrict__ out,
                           int K, int N)
{
  __shared__ float tile[32][33];
  const int tx = threadIdx.x & 31;
  const int ty = threadIdx.x >> 5;
  const int nb = blockIdx.x * 32;
  const int kb = blockIdx.y * 32;
#pragma unroll
  for (int i = 0; i < 32; i += 8)
    tile[ty + i][tx] = in[(size_t)(kb + ty + i) * N + nb + tx];
  __syncthreads();
#pragma unroll
  for (int i = 0; i < 32; i += 8)
    out[(size_t)(nb + ty + i) * K + kb + tx] = f2bf(tile[tx][ty + i]);
}

// =====================================================================
// In-place masked softmax over rows of length SK_ (1024).
// =====================================================================
__global__ __launch_bounds__(256)
void softmax_mask_kernel(float* __restrict__ P, const float* __restrict__ mask,
                         int mq, int mb, int rowsPerB)
{
  __shared__ float red[256];
  const int tid = threadIdx.x;
  const int r   = blockIdx.x;
  const int b   = r / rowsPerB;
  const int q   = r % SQ_;
  float* row = P + (size_t)r * SK_;
  const float* mrow = mask + (size_t)b * mb + (size_t)q * mq;

  float x[4];
  float mx = -3.4e38f;
#pragma unroll
  for (int i = 0; i < 4; ++i) {
    int k = tid + i * 256;
    float v = row[k] + mrow[k] * -1e9f;
    x[i] = v;
    mx = fmaxf(mx, v);
  }
  red[tid] = mx; __syncthreads();
  for (int s = 128; s > 0; s >>= 1) {
    if (tid < s) red[tid] = fmaxf(red[tid], red[tid + s]);
    __syncthreads();
  }
  mx = red[0]; __syncthreads();

  float sum = 0.f;
#pragma unroll
  for (int i = 0; i < 4; ++i) { x[i] = __expf(x[i] - mx); sum += x[i]; }
  red[tid] = sum; __syncthreads();
  for (int s = 128; s > 0; s >>= 1) {
    if (tid < s) red[tid] += red[tid + s];
    __syncthreads();
  }
  float inv = 1.f / red[0];
#pragma unroll
  for (int i = 0; i < 4; ++i) row[tid + i * 256] = x[i] * inv;
}

// =====================================================================
// Row LayerNorm over D_ (1024)
// =====================================================================
__global__ __launch_bounds__(256)
void layernorm_kernel(const float* __restrict__ X, const float* __restrict__ gamma,
                      const float* __restrict__ beta, float* __restrict__ Y)
{
  __shared__ float rs[256];
  __shared__ float rs2[256];
  const int tid = threadIdx.x;
  const int r   = blockIdx.x;
  const float* x = X + (size_t)r * D_;
  float v[4], s = 0.f, s2 = 0.f;
#pragma unroll
  for (int i = 0; i < 4; ++i) {
    v[i] = x[tid + i * 256];
    s += v[i]; s2 += v[i] * v[i];
  }
  rs[tid] = s; rs2[tid] = s2; __syncthreads();
  for (int st = 128; st > 0; st >>= 1) {
    if (tid < st) { rs[tid] += rs[tid + st]; rs2[tid] += rs2[tid + st]; }
    __syncthreads();
  }
  float mean = rs[0] * (1.f / D_);
  float var  = rs2[0] * (1.f / D_) - mean * mean;
  float rinv = rsqrtf(var + 1e-6f);
  float* y = Y + (size_t)r * D_;
#pragma unroll
  for (int i = 0; i < 4; ++i) {
    int c = tid + i * 256;
    y[c] = (v[i] - mean) * rinv * gamma[c] + beta[c];
  }
}

// =====================================================================
// Host-side orchestration
// =====================================================================
extern "C" void kernel_launch(void* const* d_in, const int* in_sizes, int n_in,
                              void* d_out, int out_size, void* d_ws, size_t ws_size,
                              hipStream_t stream)
{
  (void)in_sizes; (void)n_in; (void)out_size; (void)ws_size;
  const float* x    = (const float*)d_in[0];
  const float* enc  = (const float*)d_in[1];
  const float* lam  = (const float*)d_in[2];
  const float* pm   = (const float*)d_in[3];
  const float* wq1  = (const float*)d_in[4];  const float* bq1 = (const float*)d_in[5];
  const float* wk1  = (const float*)d_in[6];  const float* bk1 = (const float*)d_in[7];
  const float* wv1  = (const float*)d_in[8];  const float* bv1 = (const float*)d_in[9];
  const float* wo1  = (const float*)d_in[10]; const float* bo1 = (const float*)d_in[11];
  const float* wq2  = (const float*)d_in[12]; const float* bq2 = (const float*)d_in[13];
  const float* wk2  = (const float*)d_in[14]; const float* bk2 = (const float*)d_in[15];
  const float* wv2  = (const float*)d_in[16]; const float* bv2 = (const float*)d_in[17];
  const float* wo2  = (const float*)d_in[18]; const float* bo2 = (const float*)d_in[19];
  const float* wf1  = (const float*)d_in[20]; const float* bf1 = (const float*)d_in[21];
  const float* wf2  = (const float*)d_in[22]; const float* bf2 = (const float*)d_in[23];
  const float* g1   = (const float*)d_in[24]; const float* be1 = (const float*)d_in[25];
  const float* g2   = (const float*)d_in[26]; const float* be2 = (const float*)d_in[27];
  const float* g3   = (const float*)d_in[28]; const float* be3 = (const float*)d_in[29];

  float* out3 = (float*)d_out;
  float* aw1  = out3 + (size_t)B_ * SQ_ * D_;
  float* aw2  = aw1 + (size_t)B_ * H_ * SQ_ * SK_;

  const size_t SZ = (size_t)B_ * SQ_ * D_;
  float* ws     = (float*)d_ws;
  float* t_q    = ws;
  float* t_k    = ws + SZ;
  float* t_v    = ws + 2 * SZ;
  float* t_attn = ws + 3 * SZ;
  float* t_ffn1 = ws;                 // reuses region0 (16M floats)
  float* t_x    = ws + 4 * SZ;
  float* t_out1 = ws + 5 * SZ;
  float* t_out2 = ws + 6 * SZ;
  __bf16* pw   = (__bf16*)(ws + 7 * SZ);
  const size_t W1 = (size_t)D_ * D_;
  __bf16* pwq1 = pw + 0 * W1; __bf16* pwk1 = pw + 1 * W1;
  __bf16* pwv1 = pw + 2 * W1; __bf16* pwo1 = pw + 3 * W1;
  __bf16* pwq2 = pw + 4 * W1; __bf16* pwk2 = pw + 5 * W1;
  __bf16* pwv2 = pw + 6 * W1; __bf16* pwo2 = pw + 7 * W1;
  __bf16* pwf1 = pw + 8 * W1;
  __bf16* pwf2 = pw + 12 * W1;

  const dim3 blk(256);
  const int  MROWS = B_ * SQ_;
  const dim3 gProj(D_ / 128, MROWS / 128, 1);
  const dim3 gLogit(SK_ / 128, SQ_ / 128, B_ * H_);
  const dim3 gAV(1, SQ_ / 128, B_ * H_);
  const dim3 gFfn1(DFF_ / 128, MROWS / 128, 1);
  const dim3 gSoft(B_ * H_ * SQ_);
  const dim3 gLN(MROWS);
  const dim3 gTr(D_ / 32, D_ / 32);
  const dim3 gTrF1(DFF_ / 32, D_ / 32);
  const dim3 gTrF2(D_ / 32, DFF_ / 32);

  const long long sBH  = (long long)SQ_ * D_;
  const long long sAWb = (long long)H_ * SQ_ * SK_;
  const long long sAWh = (long long)SQ_ * SK_;
  const float scale = 0.125f;

  // ---------------- pack weights to bf16 [N][K] ----------------
  transpose_pack_kernel<<<gTr,   blk, 0, stream>>>(wq1, pwq1, D_, D_);
  transpose_pack_kernel<<<gTr,   blk, 0, stream>>>(wk1, pwk1, D_, D_);
  transpose_pack_kernel<<<gTr,   blk, 0, stream>>>(wv1, pwv1, D_, D_);
  transpose_pack_kernel<<<gTr,   blk, 0, stream>>>(wo1, pwo1, D_, D_);
  transpose_pack_kernel<<<gTr,   blk, 0, stream>>>(wq2, pwq2, D_, D_);
  transpose_pack_kernel<<<gTr,   blk, 0, stream>>>(wk2, pwk2, D_, D_);
  transpose_pack_kernel<<<gTr,   blk, 0, stream>>>(wv2, pwv2, D_, D_);
  transpose_pack_kernel<<<gTr,   blk, 0, stream>>>(wo2, pwo2, D_, D_);
  transpose_pack_kernel<<<gTrF1, blk, 0, stream>>>(wf1, pwf1, D_, DFF_);
  transpose_pack_kernel<<<gTrF2, blk, 0, stream>>>(wf2, pwf2, DFF_, D_);

  // ---------------- MHA 1 ----------------
  gemm_wmma_kernel<0,2,false,false,true><<<gProj, blk, 0, stream>>>(
      x, pwq1, bq1, nullptr, t_q, MROWS, D_, D_, D_, D_, D_, 0,
      1, 0,0, 0,0, 0,0, 1.f);
  gemm_wmma_kernel<0,2,false,false,true><<<gProj, blk, 0, stream>>>(
      x, pwk1, bk1, nullptr, t_k, MROWS, D_, D_, D_, D_, D_, 0,
      1, 0,0, 0,0, 0,0, 1.f);
  gemm_wmma_kernel<0,2,false,false,true><<<gProj, blk, 0, stream>>>(
      x, pwv1, bv1, nullptr, t_v, MROWS, D_, D_, D_, D_, D_, 0,
      1, 0,0, 0,0, 0,0, 1.f);
  gemm_wmma_kernel<0,1,false,false,false><<<gLogit, blk, 0, stream>>>(
      t_q, t_k, nullptr, nullptr, aw1, SQ_, SK_, DEPTH_, D_, D_, SK_, 0,
      H_, sBH, DEPTH_, sBH, DEPTH_, sAWb, sAWh, scale);
  softmax_mask_kernel<<<gSoft, blk, 0, stream>>>(aw1, lam, SQ_, 0, H_ * SQ_);
  gemm_wmma_kernel<0,0,false,true,false><<<gAV, blk, 0, stream>>>(
      aw1, t_v, nullptr, nullptr, t_attn, SQ_, DEPTH_, SK_, SK_, D_, D_, 0,
      H_, sAWb, sAWh, sBH, DEPTH_, sBH, DEPTH_, 1.f);
  gemm_wmma_kernel<0,2,true,false,true><<<gProj, blk, 0, stream>>>(
      t_attn, pwo1, bo1, x, t_x, MROWS, D_, D_, D_, D_, D_, D_,
      1, 0,0, 0,0, 0,0, 1.f);
  layernorm_kernel<<<gLN, blk, 0, stream>>>(t_x, g1, be1, t_out1);

  // ---------------- MHA 2 ----------------
  gemm_wmma_kernel<0,2,false,false,true><<<gProj, blk, 0, stream>>>(
      t_out1, pwq2, bq2, nullptr, t_q, MROWS, D_, D_, D_, D_, D_, 0,
      1, 0,0, 0,0, 0,0, 1.f);
  gemm_wmma_kernel<0,2,false,false,true><<<gProj, blk, 0, stream>>>(
      enc, pwk2, bk2, nullptr, t_k, MROWS, D_, D_, D_, D_, D_, 0,
      1, 0,0, 0,0, 0,0, 1.f);
  gemm_wmma_kernel<0,2,false,false,true><<<gProj, blk, 0, stream>>>(
      enc, pwv2, bv2, nullptr, t_v, MROWS, D_, D_, D_, D_, D_, 0,
      1, 0,0, 0,0, 0,0, 1.f);
  gemm_wmma_kernel<0,1,false,false,false><<<gLogit, blk, 0, stream>>>(
      t_q, t_k, nullptr, nullptr, aw2, SQ_, SK_, DEPTH_, D_, D_, SK_, 0,
      H_, sBH, DEPTH_, sBH, DEPTH_, sAWb, sAWh, scale);
  softmax_mask_kernel<<<gSoft, blk, 0, stream>>>(aw2, pm, 0, SK_, H_ * SQ_);
  gemm_wmma_kernel<0,0,false,true,false><<<gAV, blk, 0, stream>>>(
      aw2, t_v, nullptr, nullptr, t_attn, SQ_, DEPTH_, SK_, SK_, D_, D_, 0,
      H_, sAWb, sAWh, sBH, DEPTH_, sBH, DEPTH_, 1.f);
  gemm_wmma_kernel<0,2,true,false,true><<<gProj, blk, 0, stream>>>(
      t_attn, pwo2, bo2, t_out1, t_x, MROWS, D_, D_, D_, D_, D_, D_,
      1, 0,0, 0,0, 0,0, 1.f);
  layernorm_kernel<<<gLN, blk, 0, stream>>>(t_x, g2, be2, t_out2);

  // ---------------- FFN ----------------
  gemm_wmma_kernel<1,2,false,false,true><<<gFfn1, blk, 0, stream>>>(
      t_out2, pwf1, bf1, nullptr, t_ffn1, MROWS, DFF_, D_, D_, D_, DFF_, 0,
      1, 0,0, 0,0, 0,0, 1.f);
  gemm_wmma_kernel<0,2,true,false,true><<<gProj, blk, 0, stream>>>(
      t_ffn1, pwf2, bf2, t_out2, t_x, MROWS, D_, DFF_, DFF_, DFF_, D_, D_,
      1, 0,0, 0,0, 0,0, 1.f);
  layernorm_kernel<<<gLN, blk, 0, stream>>>(t_x, g3, be3, out3);
}
